// ResidualAttentionBlock_71159018160491
// MI455X (gfx1250) — matrix operand order repaired and checked by reading, not compiled, not run on hardware
//
#include <hip/hip_runtime.h>
#include <hip/hip_bf16.h>
#include <cstdint>
#include <cstddef>

// ---------------------------------------------------------------------------
// CDNA5 (gfx1250) implementation: bf16 WMMA (v_wmma_f32_16x16x32_bf16) for all
// GEMMs, fp32 VALU for LN / softmax / epilogues. wave32; A/B use contiguous k = 16h + s (h = lane/16).
// A-row bounds handled by CLAMPING the row index (no exec branches in the
// K-loop). All WMMA K-loops are software-pipelined (prefetch next step's
// fragments before issuing WMMAs) with unroll 1 so accumulators keep fixed
// registers and load waits are partial.
// ---------------------------------------------------------------------------

typedef __bf16 bf16_t;
typedef bf16_t bf16x16 __attribute__((ext_vector_type(16)));
typedef bf16_t bf16x8  __attribute__((ext_vector_type(8)));
typedef float  f32x8   __attribute__((ext_vector_type(8)));

#define S_LEN   577
#define S_PAD   608           // 19 * 32 (K-step for WMMA over sequence)
#define BATCH   8
#define NHEAD   16
#define HDIM    64
#define DMODEL  1024
#define DFF     4096
#define BOXN    16
#define NX      17            // box slots + 1 unmasked CLS path
#define MROWS   (BATCH * NX)  // 136
#define NROWS   (S_LEN * BATCH) // 4616 rows of x / h

__device__ __forceinline__ bf16_t to_bf16(float f) { return (bf16_t)f; }

// A fragment: 16x32 bf16 tile, row-major source with leading dim lda.
// Lane L: row mBase+(L&15); 16 contiguous K elems kb..kb+15, kb = k0 + (L>=16 ? 16 : 0).
// Row index clamped to [0, Mlim-1]: no exec divergence in the hot loop.
__device__ __forceinline__ bf16x16 load_a_frag(const bf16_t* __restrict__ A, int lda,
                                               int Mlim, int mBase, int k0, int lane) {
  int m  = mBase + (lane & 15);
  m = (m < Mlim) ? m : (Mlim - 1);
  int kb = k0 + ((lane & 16) ? 16 : 0);
  const bf16_t* p = A + (size_t)m * lda + kb;
  bf16x8 lo = *(const bf16x8*)(p);       // K = kb .. kb+7
  bf16x8 hi = *(const bf16x8*)(p + 8);   // K = kb+8 .. kb+15
  bf16x16 a;
#pragma unroll
  for (int i = 0; i < 8; ++i) { a[i] = lo[i]; a[i + 8] = hi[i]; }
  return a;
}

// B fragment: 32x16 bf16 tile from W[N,K] row-major (computes X @ W^T).
// Lane L: column nBase+(L&15); 16 contiguous K values at k0 + (L>=16 ? 16 : 0).
__device__ __forceinline__ bf16x16 load_b_frag(const bf16_t* __restrict__ W, int ldw,
                                               int nBase, int k0, int lane) {
  int n  = nBase + (lane & 15);
  int kb = k0 + ((lane & 16) ? 16 : 0);
  return *(const bf16x16*)(W + (size_t)n * ldw + kb);
}

// ---------------------------------------------------------------------------
// Generic bf16 GEMM: C[M,N] = A[M,K] @ W[N,K]^T + bias, with mode-based epilogue.
// Block = 128 threads (4 waves). Wave tile: (MF*16) rows x 64 cols; block tile
// (MF*64) x 64. MF=2 for the big GEMM (8 WMMAs per K-step sharing 4 B frags),
// MF=1 for the small M=136 GEMMs.
// mode 0: split K/V store  -> out = k[b][h][s][e] bf16, out2 = vT[b][h][e][s] bf16
// mode 1: fp32 store       -> out[row*ldc + col]
// mode 2: QuickGELU bf16   -> out[row*ldc + col]
// mode 3: + aux residual, scatter to final output (mask_cls block | cls block)
// ---------------------------------------------------------------------------
template <int MF>
__global__ __launch_bounds__(128) void gemm_bf16_nt(
    const bf16_t* __restrict__ A, const bf16_t* __restrict__ W,
    const float* __restrict__ bias, int M, int N, int K,
    int mode, int ldc, void* __restrict__ out, void* __restrict__ out2,
    const float* __restrict__ aux)
{
  (void)N;
  const int lane  = threadIdx.x & 31;
  const int wave  = threadIdx.x >> 5;
  const int mBase = (blockIdx.x * 4 + wave) * (MF * 16);
  const int nBase = blockIdx.y * 64;

  f32x8 acc[MF][4] = {};
  bf16x16 a[MF], b[4];
#pragma unroll
  for (int f = 0; f < MF; ++f) a[f] = load_a_frag(A, K, M, mBase + 16 * f, 0, lane);
#pragma unroll
  for (int t = 0; t < 4; ++t) b[t] = load_b_frag(W, K, nBase + 16 * t, 0, lane);

#pragma unroll 1
  for (int k0 = 0; k0 < K; k0 += 32) {
    // Prefetch next K-step (last iteration harmlessly refetches k=0: branch-free body)
    const int kn = (k0 + 32 < K) ? (k0 + 32) : 0;
    bf16x16 an[MF], bn[4];
#pragma unroll
    for (int f = 0; f < MF; ++f) an[f] = load_a_frag(A, K, M, mBase + 16 * f, kn, lane);
#pragma unroll
    for (int t = 0; t < 4; ++t) bn[t] = load_b_frag(W, K, nBase + 16 * t, kn, lane);
    // Compute current step
#pragma unroll
    for (int t = 0; t < 4; ++t)
#pragma unroll
      for (int f = 0; f < MF; ++f)
        acc[f][t] = __builtin_amdgcn_wmma_f32_16x16x32_bf16(false, a[f], false, b[t],
                                                            (short)0, acc[f][t], false, false);
#pragma unroll
    for (int f = 0; f < MF; ++f) a[f] = an[f];
#pragma unroll
    for (int t = 0; t < 4; ++t) b[t] = bn[t];
  }

  const int rowOff = (lane & 16) ? 8 : 0;
#pragma unroll
  for (int f = 0; f < MF; ++f) {
#pragma unroll
    for (int t = 0; t < 4; ++t) {
      int col = nBase + 16 * t + (lane & 15);
      float bv = bias ? bias[col] : 0.0f;
#pragma unroll
      for (int r = 0; r < 8; ++r) {
        int row = mBase + 16 * f + rowOff + r;
        if (row >= M) continue;
        float val = acc[f][t][r] + bv;
        if (mode == 0) {
          int s = row >> 3, bb = row & 7;              // row = s*BATCH + b
          int part = col >> 10, h = (col >> 6) & 15, e = col & 63;
          bf16_t v = to_bf16(val);
          if (part == 0)
            ((bf16_t*)out)[(((size_t)(bb * NHEAD + h) * S_PAD + s) << 6) + e] = v;   // K
          else
            ((bf16_t*)out2)[((size_t)(bb * NHEAD + h) * HDIM + e) * S_PAD + s] = v;  // V^T
        } else if (mode == 1) {
          ((float*)out)[(size_t)row * ldc + col] = val;
        } else if (mode == 2) {
          float g = val * (1.0f / (1.0f + __expf(-1.702f * val)));                   // QuickGELU
          ((bf16_t*)out)[(size_t)row * ldc + col] = to_bf16(g);
        } else {                                                                     // mode 3
          val += aux[(size_t)row * DMODEL + col];
          int bb = row / NX, xi = row % NX;
          size_t idx = (xi == 0)
              ? (size_t)(BATCH * BOXN * DMODEL) + (size_t)bb * DMODEL + col          // cls
              : ((size_t)(bb * BOXN + (xi - 1))) * DMODEL + col;                     // mask_cls
          ((float*)out)[idx] = val;
        }
      }
    }
  }
}

// ---------------------------------------------------------------------------
// Batched P·V: per (b,h) block, out[17,64] = P[17,S_PAD] @ V^T[64,S_PAD]^T.
// 4 waves: wave&1 -> M tile (0/16), wave>>1 -> 32-col half; 2 WMMAs per K-step.
// Software-pipelined like the main GEMM.
// ---------------------------------------------------------------------------
__global__ __launch_bounds__(128) void attn_pv(const bf16_t* __restrict__ p,
                                               const bf16_t* __restrict__ vT,
                                               bf16_t* __restrict__ out_cat)
{
  const int lane = threadIdx.x & 31;
  const int wave = threadIdx.x >> 5;
  const int bh = blockIdx.x;            // b*16 + h
  const int b = bh >> 4, h = bh & 15;
  const bf16_t* Ap = p  + (size_t)(b * NX) * (NHEAD * S_PAD) + (size_t)h * S_PAD;
  const bf16_t* Bp = vT + (size_t)bh * HDIM * S_PAD;
  const int mBase = (wave & 1) * 16;
  const int nBase = (wave >> 1) * 32;
  const int lda = NHEAD * S_PAD;

  f32x8 acc[2] = {};
  bf16x16 a = load_a_frag(Ap, lda, NX, mBase, 0, lane);
  bf16x16 b0 = load_b_frag(Bp, S_PAD, nBase, 0, lane);
  bf16x16 b1 = load_b_frag(Bp, S_PAD, nBase + 16, 0, lane);

#pragma unroll 1
  for (int k0 = 0; k0 < S_PAD; k0 += 32) {
    const int kn = (k0 + 32 < S_PAD) ? (k0 + 32) : 0;
    bf16x16 an  = load_a_frag(Ap, lda, NX, mBase, kn, lane);
    bf16x16 b0n = load_b_frag(Bp, S_PAD, nBase, kn, lane);
    bf16x16 b1n = load_b_frag(Bp, S_PAD, nBase + 16, kn, lane);
    acc[0] = __builtin_amdgcn_wmma_f32_16x16x32_bf16(false, a, false, b0,
                                                     (short)0, acc[0], false, false);
    acc[1] = __builtin_amdgcn_wmma_f32_16x16x32_bf16(false, a, false, b1,
                                                     (short)0, acc[1], false, false);
    a = an; b0 = b0n; b1 = b1n;
  }

  const int rowOff = (lane & 16) ? 8 : 0;
  const int e0 = nBase + (lane & 15);
#pragma unroll
  for (int t = 0; t < 2; ++t) {
#pragma unroll
    for (int r = 0; r < 8; ++r) {
      int xi = mBase + rowOff + r;
      if (xi < NX)
        out_cat[(size_t)(b * NX + xi) * DMODEL + h * HDIM + e0 + 16 * t] = to_bf16(acc[t][r]);
    }
  }
}

// ---------------------------------------------------------------------------
// LayerNorm over D=1024 -> bf16 (one block per row)
// ---------------------------------------------------------------------------
__global__ __launch_bounds__(256) void ln_to_bf16(const float* __restrict__ x,
                                                  const float* __restrict__ g,
                                                  const float* __restrict__ bta,
                                                  bf16_t* __restrict__ out)
{
  __shared__ float red[256];
  const int r = blockIdx.x, tid = threadIdx.x;
  const float* xr = x + (size_t)r * DMODEL;
  float s = 0.f;
  for (int i = tid; i < DMODEL; i += 256) s += xr[i];
  red[tid] = s; __syncthreads();
  for (int o = 128; o > 0; o >>= 1) { if (tid < o) red[tid] += red[tid + o]; __syncthreads(); }
  float mean = red[0] * (1.0f / DMODEL); __syncthreads();
  float vs = 0.f;
  for (int i = tid; i < DMODEL; i += 256) { float d = xr[i] - mean; vs += d * d; }
  red[tid] = vs; __syncthreads();
  for (int o = 128; o > 0; o >>= 1) { if (tid < o) red[tid] += red[tid + o]; __syncthreads(); }
  float rstd = rsqrtf(red[0] * (1.0f / DMODEL) + 1e-5f);
  for (int i = tid; i < DMODEL; i += 256)
    out[(size_t)r * DMODEL + i] = to_bf16((xr[i] - mean) * rstd * g[i] + bta[i]);
}

// Residual (x0[b] + attn_out) -> t (fp32, kept for final residual) and LN2 -> u (bf16)
__global__ __launch_bounds__(256) void resid_ln(const float* __restrict__ x0,
                                                const float* __restrict__ ao,
                                                const float* __restrict__ g,
                                                const float* __restrict__ bta,
                                                float* __restrict__ t,
                                                bf16_t* __restrict__ u)
{
  __shared__ float red[256];
  const int r = blockIdx.x, tid = threadIdx.x;
  const int b = r / NX;
  const float* xr = x0 + (size_t)b * DMODEL;
  const float* ar = ao + (size_t)r * DMODEL;
  float loc[4];
  float s = 0.f;
  for (int i = tid, j = 0; i < DMODEL; i += 256, ++j) {
    float v = xr[i] + ar[i];
    loc[j] = v;
    t[(size_t)r * DMODEL + i] = v;
    s += v;
  }
  red[tid] = s; __syncthreads();
  for (int o = 128; o > 0; o >>= 1) { if (tid < o) red[tid] += red[tid + o]; __syncthreads(); }
  float mean = red[0] * (1.0f / DMODEL); __syncthreads();
  float vs = 0.f;
  for (int i = tid, j = 0; i < DMODEL; i += 256, ++j) { float d = loc[j] - mean; vs += d * d; }
  red[tid] = vs; __syncthreads();
  for (int o = 128; o > 0; o >>= 1) { if (tid < o) red[tid] += red[tid + o]; __syncthreads(); }
  float rstd = rsqrtf(red[0] * (1.0f / DMODEL) + 1e-5f);
  for (int i = tid, j = 0; i < DMODEL; i += 256, ++j)
    u[(size_t)r * DMODEL + i] = to_bf16((loc[j] - mean) * rstd * g[i] + bta[i]);
}

// q0[b, h*64+e] = (LN(x[0,b]) @ Wq^T + bq) / 8
__global__ __launch_bounds__(256) void q0_kernel(const bf16_t* __restrict__ hbf,
                                                 const bf16_t* __restrict__ wq,
                                                 const float* __restrict__ bq,
                                                 float* __restrict__ q0)
{
  int t = blockIdx.x * 256 + threadIdx.x;
  if (t >= BATCH * DMODEL) return;
  int b = t >> 10, n = t & 1023;
  const bf16_t* hr = hbf + (size_t)b * DMODEL;   // s=0 rows are rows 0..7 of h
  const bf16_t* wr = wq + (size_t)n * DMODEL;
  float acc = bq[n];
#pragma unroll 8
  for (int k = 0; k < DMODEL; ++k) acc += (float)hr[k] * (float)wr[k];
  q0[t] = acc * 0.125f;
}

// attn[bh, s] = q0[bh,:] . k[bh, s, :]
__global__ __launch_bounds__(256) void attn_dot(const float* __restrict__ q0,
                                                const bf16_t* __restrict__ kbuf,
                                                float* __restrict__ attn)
{
  int t = blockIdx.x * 256 + threadIdx.x;
  if (t >= BATCH * NHEAD * S_LEN) return;
  int s = t % S_LEN, bh = t / S_LEN;
  const float* q = q0 + (size_t)bh * HDIM;
  const bf16_t* kk = kbuf + ((size_t)bh * S_PAD + s) * HDIM;
  float acc = 0.f;
#pragma unroll 8
  for (int e = 0; e < HDIM; ++e) acc += q[e] * (float)kk[e];
  attn[(size_t)bh * S_PAD + s] = acc;
}

// softmax over S for 17 variants per (b,h): xi==0 plain, xi>=1 with box mask.
// p layout: [b][xi][h][S_PAD] bf16; tail s in [577,608) zeroed.
__global__ __launch_bounds__(256) void softmax_p(const float* __restrict__ attn,
                                                 const float* __restrict__ mask,
                                                 bf16_t* __restrict__ p)
{
  __shared__ float red[256];
  const int id = blockIdx.x;                 // (b*NX + xi)*NHEAD + h
  const int h = id % NHEAD, bxi = id / NHEAD;
  const int xi = bxi % NX, b = bxi / NX;
  const int tid = threadIdx.x;
  const float* arow = attn + (size_t)(b * NHEAD + h) * S_PAD;
  const float* mrow = (xi == 0) ? nullptr
      : mask + (size_t)((b * BOXN + (xi - 1)) * NHEAD + h) * S_LEN;
  float loc[3];
  float mx = -3.0e38f;
  for (int s = tid, j = 0; s < S_LEN; s += 256, ++j) {
    float v = arow[s] + (mrow ? mrow[s] : 0.f);
    loc[j] = v;
    mx = fmaxf(mx, v);
  }
  red[tid] = mx; __syncthreads();
  for (int o = 128; o > 0; o >>= 1) { if (tid < o) red[tid] = fmaxf(red[tid], red[tid + o]); __syncthreads(); }
  float M = red[0]; __syncthreads();
  float sum = 0.f;
  for (int s = tid, j = 0; s < S_LEN; s += 256, ++j) { loc[j] = __expf(loc[j] - M); sum += loc[j]; }
  red[tid] = sum; __syncthreads();
  for (int o = 128; o > 0; o >>= 1) { if (tid < o) red[tid] += red[tid + o]; __syncthreads(); }
  float inv = 1.0f / red[0];
  bf16_t* prow = p + (size_t)id * S_PAD;
  for (int s = tid, j = 0; s < S_LEN; s += 256, ++j) prow[s] = to_bf16(loc[j] * inv);
  for (int s = S_LEN + tid; s < S_PAD; s += 256) prow[s] = (bf16_t)0.0f;
}

__global__ __launch_bounds__(256) void f32_to_bf16_k(const float* __restrict__ in,
                                                     bf16_t* __restrict__ out, int n)
{
  int i = blockIdx.x * 256 + threadIdx.x;
  int stride = gridDim.x * 256;
  for (; i < n; i += stride) out[i] = to_bf16(in[i]);
}

// ---------------------------------------------------------------------------
extern "C" void kernel_launch(void* const* d_in, const int* in_sizes, int n_in,
                              void* d_out, int out_size, void* d_ws, size_t ws_size,
                              hipStream_t stream)
{
  (void)in_sizes; (void)n_in; (void)out_size; (void)ws_size;
  const float* x    = (const float*)d_in[0];   // [577,8,1024]
  const float* mask = (const float*)d_in[1];   // [2048,1,577]
  const float* ipw  = (const float*)d_in[5];   // [3072,1024]
  const float* ipb  = (const float*)d_in[6];
  const float* outw = (const float*)d_in[7];   // [1024,1024]
  const float* outb = (const float*)d_in[8];
  const float* ln1g = (const float*)d_in[9];
  const float* ln1b = (const float*)d_in[10];
  const float* ln2g = (const float*)d_in[11];
  const float* ln2b = (const float*)d_in[12];
  const float* fcw  = (const float*)d_in[13];  // [4096,1024]
  const float* fcb  = (const float*)d_in[14];
  const float* pjw  = (const float*)d_in[15];  // [1024,4096]
  const float* pjb  = (const float*)d_in[16];

  char* ws = (char*)d_ws;
  size_t off = 0;
  auto alloc = [&](size_t bytes) -> char* {
    char* p = ws + off;
    off += (bytes + 255) & ~(size_t)255;
    return p;
  };
  bf16_t* w_ip   = (bf16_t*)alloc((size_t)3072 * 1024 * 2);
  bf16_t* w_out  = (bf16_t*)alloc((size_t)1024 * 1024 * 2);
  bf16_t* w_fc   = (bf16_t*)alloc((size_t)4096 * 1024 * 2);
  bf16_t* w_pj   = (bf16_t*)alloc((size_t)4096 * 1024 * 2);
  bf16_t* hbf    = (bf16_t*)alloc((size_t)NROWS * DMODEL * 2);
  bf16_t* kbuf   = (bf16_t*)alloc((size_t)BATCH * NHEAD * S_PAD * HDIM * 2);
  const size_t vt_bytes = (size_t)BATCH * NHEAD * HDIM * S_PAD * 2;
  bf16_t* vtbuf  = (bf16_t*)alloc(vt_bytes);
  float*  q0     = (float*)alloc((size_t)BATCH * DMODEL * 4);
  float*  attn   = (float*)alloc((size_t)BATCH * NHEAD * S_PAD * 4);
  bf16_t* pbuf   = (bf16_t*)alloc((size_t)BATCH * NX * NHEAD * S_PAD * 2);
  bf16_t* ocat   = (bf16_t*)alloc((size_t)MROWS * DMODEL * 2);
  float*  aout   = (float*)alloc((size_t)MROWS * DMODEL * 4);
  float*  tbuf   = (float*)alloc((size_t)MROWS * DMODEL * 4);
  bf16_t* ubuf   = (bf16_t*)alloc((size_t)MROWS * DMODEL * 2);
  bf16_t* mbuf   = (bf16_t*)alloc((size_t)MROWS * DFF * 2);

  // Zero V^T so padded s in [577,608) contributes nothing to P·V.
  hipMemsetAsync(vtbuf, 0, vt_bytes, stream);

  // fp32 -> bf16 weight conversion (each call; no cross-call caching allowed)
  f32_to_bf16_k<<<4096, 256, 0, stream>>>(ipw, w_ip, 3072 * 1024);
  f32_to_bf16_k<<<4096, 256, 0, stream>>>(outw, w_out, 1024 * 1024);
  f32_to_bf16_k<<<4096, 256, 0, stream>>>(fcw, w_fc, 4096 * 1024);
  f32_to_bf16_k<<<4096, 256, 0, stream>>>(pjw, w_pj, 4096 * 1024);

  // LN1 over all [S*B] rows -> bf16 h
  ln_to_bf16<<<NROWS, 256, 0, stream>>>(x, ln1g, ln1b, hbf);

  // K/V projection: h[4616,1024] @ W_kv[2048,1024]^T  (WMMA, 19.4 GFLOP)
  // MF=2: block tile 128x64, 8 WMMAs per K-step sharing 4 B fragments.
  gemm_bf16_nt<2><<<dim3((NROWS + 127) / 128, 2048 / 64), 128, 0, stream>>>(
      hbf, w_ip + (size_t)DMODEL * DMODEL, ipb + DMODEL,
      NROWS, 2048, DMODEL, /*mode=*/0, 0, kbuf, vtbuf, nullptr);

  // q0 (CLS rows only) + attention scores
  q0_kernel<<<(BATCH * DMODEL + 255) / 256, 256, 0, stream>>>(hbf, w_ip, ipb, q0);
  attn_dot<<<(BATCH * NHEAD * S_LEN + 255) / 256, 256, 0, stream>>>(q0, kbuf, attn);

  // 2176 softmax rows: plain (xi=0) + 16 masked variants
  softmax_p<<<BATCH * NX * NHEAD, 256, 0, stream>>>(attn, mask, pbuf);

  // P·V (batched WMMA over heads) -> out_cat[136,1024] bf16
  attn_pv<<<BATCH * NHEAD, 128, 0, stream>>>(pbuf, vtbuf, ocat);

  // out projection: out_cat @ out_w^T + out_b -> fp32 attn_out
  gemm_bf16_nt<1><<<dim3((MROWS + 63) / 64, DMODEL / 64), 128, 0, stream>>>(
      ocat, w_out, outb, MROWS, DMODEL, DMODEL, /*mode=*/1, DMODEL, aout, nullptr, nullptr);

  // residual + LN2
  resid_ln<<<MROWS, 256, 0, stream>>>(x, aout, ln2g, ln2b, tbuf, ubuf);

  // MLP fc + QuickGELU -> bf16 m
  gemm_bf16_nt<1><<<dim3((MROWS + 63) / 64, DFF / 64), 128, 0, stream>>>(
      ubuf, w_fc, fcb, MROWS, DFF, DMODEL, /*mode=*/2, DFF, mbuf, nullptr, nullptr);

  // MLP proj + residual t, scatter into d_out (mask_cls[128,1024] then cls[8,1024])
  gemm_bf16_nt<1><<<dim3((MROWS + 63) / 64, DMODEL / 64), 128, 0, stream>>>(
      mbuf, w_pj, pjb, MROWS, DMODEL, DFF, /*mode=*/3, DMODEL, d_out, nullptr, tbuf);
}
